// DSQGBlockV6Physics_9483287789633
// MI455X (gfx1250) — matrix operand order repaired and checked by reading, not compiled, not run on hardware
//
#include <hip/hip_runtime.h>

// ---------------- constants (fixed problem shape) ----------------
#define BB 2
#define NN 2048
#define DD 1024
#define HH 16
#define HD 64
#define FFN 4096
#define MTOK (BB*NN)   // 4096 tokens

typedef __attribute__((ext_vector_type(16))) __bf16 v16bf;
typedef __attribute__((ext_vector_type(8)))  float  v8f;
typedef __attribute__((ext_vector_type(4)))  unsigned v4u;
typedef __attribute__((ext_vector_type(8)))  int  v8i;
typedef __attribute__((ext_vector_type(4)))  int  v4i;

#if __has_builtin(__builtin_amdgcn_tensor_load_to_lds)
#define HAS_TDM 1
#else
#define HAS_TDM 0
#endif

static __device__ __forceinline__ unsigned short f2bf(float f) {
  unsigned u = __float_as_uint(f);
  unsigned r = (u + 0x7FFFu + ((u >> 16) & 1u)) >> 16;
  return (unsigned short)r;
}

// LDS byte address of a generic pointer to __shared__ (low 32 bits of flat addr)
static __device__ __forceinline__ unsigned lds_addr_of(const void* p) {
  return (unsigned)(unsigned long long)p;
}

// ---- WMMA fragment gathers per CDNA5 ISA 7.12.2 layouts (bf16, 16x16x32) ----
static __device__ __forceinline__ v16bf load_frag_a(const unsigned short* base, int ldm, int lane) {
  int r = lane & 15, h = lane >> 4;
  v16bf a;
  unsigned* ap = (unsigned*)&a;
#pragma unroll
  for (int j = 0; j < 8; ++j) {
    int k0 = ((j >> 2) << 4) + (h << 3) + ((j & 3) << 1);
    ap[j] = *(const unsigned*)(base + r * ldm + k0);
  }
  return a;
}
// B 32x16 gathered from [n][k] storage.
static __device__ __forceinline__ v16bf load_frag_bt(const unsigned short* baseT, int ldk, int lane) {
  int r = lane & 15, h = lane >> 4;
  v16bf b;
  unsigned* bp = (unsigned*)&b;
#pragma unroll
  for (int j = 0; j < 8; ++j) {
    int k0 = ((j >> 2) << 4) + (h << 3) + ((j & 3) << 1);
    bp[j] = *(const unsigned*)(baseT + r * ldk + k0);
  }
  return b;
}
static __device__ __forceinline__ v8f wmma_bf16(v16bf a, v16bf b, v8f c) {
  return __builtin_amdgcn_wmma_f32_16x16x32_bf16(false, a, false, b, (short)0, c, false, false);
}

// ---- async global->LDS copy of one 16-byte chunk (ASYNCcnt-tracked) ----
static __device__ __forceinline__ void async_b128(unsigned lds_off, unsigned gbyte_off,
                                                  const void* base) {
  asm volatile("global_load_async_to_lds_b128 %0, %1, %2"
               :: "v"(lds_off), "v"(gbyte_off), "s"((unsigned long long)base)
               : "memory");
}
static __device__ __forceinline__ void wait_async_pending8() {
  asm volatile("s_wait_asynccnt 0x8" ::: "memory");
}
static __device__ __forceinline__ void wait_async_pending4() {
  asm volatile("s_wait_asynccnt 0x4" ::: "memory");
}
static __device__ __forceinline__ void wait_async_zero() {
  asm volatile("s_wait_asynccnt 0x0" ::: "memory");
}

#if HAS_TDM
// ---- TDM: DMA a 32(k) x 128(rows) bf16 tile, row stride = rowlen elems ----
// D# per CDNA5 ISA 8.3/8.4 (group0: count/lds/global/type; group1: dims/strides)
static __device__ __forceinline__ void tdm_load_tile(unsigned lds_addr,
                                                     const unsigned short* gbase,
                                                     long long elem_off,
                                                     int rowlen, int nrows_total) {
  unsigned long long ga = (unsigned long long)gbase + (unsigned long long)elem_off * 2ull;
  v4u g0;
  g0[0] = 1u;                                       // count=1, user desc
  g0[1] = lds_addr;                                 // lds_addr
  g0[2] = (unsigned)(ga & 0xffffffffu);             // global_addr[31:0]
  g0[3] = (unsigned)((ga >> 32) & 0x1ffffffu) | (2u << 30);  // global_addr[56:32] | type=2
  v8i g1;
  g1[0] = (int)(1u << 16);                          // workgroup_mask=0, data_size=1 (2 bytes)
  g1[1] = (int)(((unsigned)rowlen & 0xffffu) << 16);            // tensor_dim0[15:0]
  g1[2] = (int)((((unsigned)rowlen >> 16) & 0xffffu) |
                (((unsigned)nrows_total & 0xffffu) << 16));     // td0[31:16] | td1[15:0]
  g1[3] = (int)((((unsigned)nrows_total >> 16) & 0xffffu) |
                (32u << 16));                                   // td1[31:16] | tile_dim0=32
  g1[4] = (int)(128u);                              // tile_dim1=128, tile_dim2=0
  g1[5] = (int)rowlen;                              // tensor_dim0_stride[31:0]
  g1[6] = 0;                                        // stride0 hi, stride1 lo16
  g1[7] = 0;
  v4i z = {0, 0, 0, 0};
#if __clang_major__ >= 23
  v8i z8 = {0, 0, 0, 0, 0, 0, 0, 0};
  __builtin_amdgcn_tensor_load_to_lds(g0, g1, z, z, z8, 0);
#else
  __builtin_amdgcn_tensor_load_to_lds(g0, g1, z, z, 0);
#endif
}
#endif

// ---------------- f32 -> bf16 transposed weight conversion ----------------
// src: [K][N] row-major f32 ; dst: [N][K] bf16
__global__ void cvt_t_kernel(const float* __restrict__ src, unsigned short* __restrict__ dst,
                             int K, int N) {
  int i = blockIdx.x * blockDim.x + threadIdx.x;
  if (i < K * N) {
    int k = i / N, n = i - k * N;
    dst[(size_t)n * K + k] = f2bf(src[i]);
  }
}

// ---------------- LayerNorm: f32 in -> bf16 out (+ optional f32 out) ----------------
__global__ __launch_bounds__(256) void ln_kernel(const float* __restrict__ x,
                                                 const float* __restrict__ g,
                                                 const float* __restrict__ bta,
                                                 unsigned short* __restrict__ out_bf,
                                                 float* __restrict__ out_f) {
  __shared__ float r1[256], r2[256];
  int tok = blockIdx.x, tid = threadIdx.x;
  const float* xp = x + (size_t)tok * DD;
  float s = 0.f, s2 = 0.f;
  for (int i = tid; i < DD; i += 256) { float v = xp[i]; s += v; s2 += v * v; }
  r1[tid] = s; r2[tid] = s2; __syncthreads();
  for (int st = 128; st > 0; st >>= 1) {
    if (tid < st) { r1[tid] += r1[tid + st]; r2[tid] += r2[tid + st]; }
    __syncthreads();
  }
  float mean = r1[0] / DD;
  float var  = r2[0] / DD - mean * mean;
  float inv  = rsqrtf(var + 1e-5f);
  for (int i = tid; i < DD; i += 256) {
    float y = (xp[i] - mean) * inv * g[i] + bta[i];
    out_bf[(size_t)tok * DD + i] = f2bf(y);
    if (out_f) out_f[(size_t)tok * DD + i] = y;
  }
}

// ---------------- causal EMA scan: one thread per (b,d) channel ----------------
__global__ void ema_kernel(const float* __restrict__ xi, const float* __restrict__ ema_factor,
                           float* __restrict__ pool) {
  int idx = blockIdx.x * blockDim.x + threadIdx.x;
  if (idx >= BB * DD) return;
  int b = idx >> 10, d = idx & (DD - 1);
  float a = fminf(fmaxf(ema_factor[0], 1e-5f), 1.0f);
  float om = 1.f - a;
  float p = 0.f;
  const float* xp = xi + (size_t)b * NN * DD + d;
  float* pp = pool + (size_t)b * NN * DD + d;
  for (int t = 0; t < NN; ++t) { p = om * p + a * xp[(size_t)t * DD]; pp[(size_t)t * DD] = p; }
}

// ---------------- AGC: per-token RMS normalize in place ----------------
__global__ __launch_bounds__(256) void agc_kernel(float* __restrict__ pool) {
  __shared__ float red[256];
  int tok = blockIdx.x, tid = threadIdx.x;
  float* p = pool + (size_t)tok * DD;
  float ss = 0.f;
  for (int i = tid; i < DD; i += 256) { float v = p[i]; ss += v * v; }
  red[tid] = ss; __syncthreads();
  for (int st = 128; st > 0; st >>= 1) { if (tid < st) red[tid] += red[tid + st]; __syncthreads(); }
  float rms = sqrtf(red[0] / (float)DD);
  float inv = 1.f / (rms + 1e-6f);
  for (int i = tid; i < DD; i += 256) p[i] *= inv;
}

// ---------------- tiled bf16 WMMA GEMM, TDM/async double-buffered staging ----------------
// out = epilogue(A[M,K] @ WT[Nc,K]^T + bias)
// MODE: 1 sigmoid*pool->bf16 tok | 2 bias->f32 tok | 3 bias+add->bf16 HEAD [B,H,N,HD]
//       4 *0.125->bf16 HEAD | 5 bias+add->f32 tok | 6 gelu->bf16 tok | 7 bias+add->bf16 HEAD^T [B,H,HD,N]
#define GBM 128
#define GBN 128
#define GBK 32
template <int MODE>
__global__ __launch_bounds__(256) void gemm_bf16_kernel(
    const unsigned short* __restrict__ A, const unsigned short* __restrict__ WT,
    const float* __restrict__ bias, const float* __restrict__ addb,
    const float* __restrict__ poolb,
    unsigned short* __restrict__ outb, float* __restrict__ outf,
    int M, int K, int Nc) {
  __shared__ unsigned short As[2][GBM * GBK];   // [m][k]
  __shared__ unsigned short Bs[2][GBN * GBK];   // [n][k]
  int tid = threadIdx.x;
  int lane = tid & 31, w = tid >> 5;
  int bm = blockIdx.y * GBM, bn = blockIdx.x * GBN;
  int wm = (w >> 1) * 32;
  int wn = (w & 1) * 64;

  v8f acc[2][4];
#pragma unroll
  for (int mi = 0; mi < 2; ++mi)
#pragma unroll
    for (int nf = 0; nf < 4; ++nf)
#pragma unroll
      for (int e = 0; e < 8; ++e) acc[mi][nf][e] = 0.f;

  auto issueTile = [&](int buf, int k0) {
#if HAS_TDM
    if (w == 0) {
      tdm_load_tile(lds_addr_of(&As[buf][0]), A,  (long long)bm * K + k0, K, M);
      tdm_load_tile(lds_addr_of(&Bs[buf][0]), WT, (long long)bn * K + k0, K, Nc);
    }
#else
#pragma unroll
    for (int i = 0; i < 2; ++i) {
      int id = tid + i * 256;                 // 0..511
      int row = id >> 2, c8 = (id & 3) << 3;  // 128 rows x 32 cols (8 bf16 / chunk)
      async_b128(lds_addr_of(&As[buf][row * GBK + c8]),
                 (unsigned)(((size_t)(bm + row) * K + k0 + c8) * 2), A);
      async_b128(lds_addr_of(&Bs[buf][row * GBK + c8]),
                 (unsigned)(((size_t)(bn + row) * K + k0 + c8) * 2), WT);
    }
#endif
  };

  issueTile(0, 0);
  int nk = K / GBK;
  for (int it = 0; it < nk; ++it) {
    int buf = it & 1;
    bool pre = (it + 1 < nk);
    if (pre) issueTile(buf ^ 1, (it + 1) * GBK);
#if HAS_TDM
    if (w == 0) {
      if (pre) __builtin_amdgcn_s_wait_tensorcnt(2);
      else     __builtin_amdgcn_s_wait_tensorcnt(0);
    }
#else
    if (pre) wait_async_pending4(); else wait_async_zero();
#endif
    __syncthreads();
    v16bf af[2];
    af[0] = load_frag_a(&As[buf][(wm + 0)  * GBK], GBK, lane);
    af[1] = load_frag_a(&As[buf][(wm + 16) * GBK], GBK, lane);
    v16bf bfv[4];
#pragma unroll
    for (int nf = 0; nf < 4; ++nf) bfv[nf] = load_frag_bt(&Bs[buf][(wn + nf * 16) * GBK], GBK, lane);
#pragma unroll
    for (int mi = 0; mi < 2; ++mi)
#pragma unroll
      for (int nf = 0; nf < 4; ++nf) acc[mi][nf] = wmma_bf16(af[mi], bfv[nf], acc[mi][nf]);
    __syncthreads();
  }

  // epilogue (specialized at compile time)
  int r = lane & 15, hh = lane >> 4;
#pragma unroll
  for (int mi = 0; mi < 2; ++mi)
#pragma unroll
    for (int nf = 0; nf < 4; ++nf)
#pragma unroll
      for (int i = 0; i < 8; ++i) {
        int row = bm + wm + mi * 16 + hh * 8 + i;
        int col = bn + wn + nf * 16 + r;
        size_t off = (size_t)row * Nc + col;
        float v = acc[mi][nf][i] + bias[col];
        if constexpr (MODE == 1) {
          float s = 1.f / (1.f + __expf(-v));
          outb[off] = f2bf(s * poolb[off]);
        } else if constexpr (MODE == 2) {
          outf[off] = v;
        } else if constexpr (MODE == 3) {
          float rv = v + addb[off];
          int h = col >> 6, d = col & 63;
          int b = row >> 11, t = row & (NN - 1);
          outb[(((size_t)(b * HH + h)) * NN + t) * HD + d] = f2bf(rv);
        } else if constexpr (MODE == 4) {
          float rv = v * 0.125f;
          int h = col >> 6, d = col & 63;
          int b = row >> 11, t = row & (NN - 1);
          outb[(((size_t)(b * HH + h)) * NN + t) * HD + d] = f2bf(rv);
        } else if constexpr (MODE == 5) {
          outf[off] = v + addb[off];
        } else if constexpr (MODE == 6) {
          float t = tanhf(0.7978845608f * (v + 0.044715f * v * v * v));
          outb[off] = f2bf(0.5f * v * (1.f + t));
        } else { // 7: transposed head layout [B,H,HD,N]
          float rv = v + addb[off];
          int h = col >> 6, d = col & 63;
          int b = row >> 11, t = row & (NN - 1);
          outb[(((size_t)(b * HH + h)) * HD + d) * NN + t] = f2bf(rv);
        }
      }
}

// ---------------- causal flash attention (bf16 WMMA, async-staged K/V) ----------------
// q,k: [B,H,N,HD] bf16 (q pre-scaled); vT: [B,H,HD,N] bf16; o_tok: [B,N,D] bf16
__global__ __launch_bounds__(128) void attn_kernel(const unsigned short* __restrict__ q,
                                                   const unsigned short* __restrict__ kk,
                                                   const unsigned short* __restrict__ vT,
                                                   unsigned short* __restrict__ o_tok) {
  __shared__ unsigned short Ks[2][64 * 64];     // [key][hd]
  __shared__ unsigned short Vs[2][64 * 64];     // [hd][key]
  __shared__ unsigned short Ps[4][16 * 64];
  __shared__ float Ss[4][16 * 64];
  __shared__ float mrow[64], lrow[64], srow[64];
  __shared__ float red[64][2];

  int qb = blockIdx.x, h = blockIdx.y, b = blockIdx.z;
  int tid = threadIdx.x, lane = tid & 31, w = tid >> 5;
  size_t headOff = ((size_t)(b * HH + h)) * NN * HD;

  const unsigned short* Qp = q + headOff + (size_t)(qb * 64 + w * 16) * HD;
  v16bf qf[2];
  qf[0] = load_frag_a(Qp + 0,  HD, lane);
  qf[1] = load_frag_a(Qp + 32, HD, lane);

  v8f of[4];
#pragma unroll
  for (int nf = 0; nf < 4; ++nf)
#pragma unroll
    for (int e = 0; e < 8; ++e) of[nf][e] = 0.f;
  if (tid < 64) { mrow[tid] = -3e38f; lrow[tid] = 0.f; }

  auto issueKV = [&](int buf, int kb) {
#pragma unroll
    for (int i = 0; i < 4; ++i) {
      int id = tid + i * 128;                 // 0..511
      int row = id >> 3, c8 = (id & 7) << 3;  // 64 rows x 64 cols
      async_b128(lds_addr_of(&Ks[buf][row * 64 + c8]),
                 (unsigned)((headOff + (size_t)(kb * 64 + row) * HD + c8) * 2), kk);
      async_b128(lds_addr_of(&Vs[buf][row * 64 + c8]),
                 (unsigned)(((((size_t)(b * HH + h)) * HD + row) * NN + kb * 64 + c8) * 2), vT);
    }
  };

  issueKV(0, 0);
  int rr = lane & 15, hh = lane >> 4;
  for (int kb = 0; kb <= qb; ++kb) {
    int buf = kb & 1;
    bool pre = (kb < qb);
    if (pre) issueKV(buf ^ 1, kb + 1);
    if (pre) wait_async_pending8(); else wait_async_zero();
    __syncthreads();

    // S = Q (16x64) . K^T
#pragma unroll
    for (int nf = 0; nf < 4; ++nf) {
      v8f s;
#pragma unroll
      for (int e = 0; e < 8; ++e) s[e] = 0.f;
      v16bf b0 = load_frag_bt(&Ks[buf][(nf * 16) * 64 + 0],  64, lane);
      v16bf b1 = load_frag_bt(&Ks[buf][(nf * 16) * 64 + 32], 64, lane);
      s = wmma_bf16(qf[0], b0, s);
      s = wmma_bf16(qf[1], b1, s);
#pragma unroll
      for (int i = 0; i < 8; ++i)
        Ss[w][(hh * 8 + i) * 64 + nf * 16 + rr] = s[i];
    }
    __syncthreads();

    // online softmax: thread -> (row = tid/2, half = tid&1)
    int row = tid >> 1, part = tid & 1;
    int wrow = row >> 4, lr = row & 15;
    int grow = qb * 64 + row;
    bool diag = (kb == qb);
    float* srowp = &Ss[wrow][lr * 64];
    float pmax = -3e38f;
    for (int c = part * 32; c < part * 32 + 32; ++c) {
      int gc = kb * 64 + c;
      float sv = (diag && gc > grow) ? -3e38f : srowp[c];
      pmax = fmaxf(pmax, sv);
    }
    red[row][part] = pmax;
    __syncthreads();
    float m_old = mrow[row];
    float m_new = fmaxf(m_old, fmaxf(red[row][0], red[row][1]));
    float scale = __expf(m_old - m_new);
    __syncthreads();
    float psum = 0.f;
    for (int c = part * 32; c < part * 32 + 32; ++c) {
      int gc = kb * 64 + c;
      float p = (diag && gc > grow) ? 0.f : __expf(srowp[c] - m_new);
      Ps[wrow][lr * 64 + c] = f2bf(p);
      psum += p;
    }
    red[row][part] = psum;
    __syncthreads();
    if (part == 0) {
      mrow[row] = m_new;
      lrow[row] = lrow[row] * scale + red[row][0] + red[row][1];
      srow[row] = scale;
    }
    __syncthreads();

    // O = O*scale + P @ V
    v16bf pf0 = load_frag_a(&Ps[w][0],  64, lane);
    v16bf pf1 = load_frag_a(&Ps[w][32], 64, lane);
#pragma unroll
    for (int nf = 0; nf < 4; ++nf) {
#pragma unroll
      for (int i = 0; i < 8; ++i) of[nf][i] *= srow[w * 16 + hh * 8 + i];
      v16bf b0 = load_frag_bt(&Vs[buf][(nf * 16) * 64 + 0],  64, lane);
      v16bf b1 = load_frag_bt(&Vs[buf][(nf * 16) * 64 + 32], 64, lane);
      of[nf] = wmma_bf16(pf0, b0, of[nf]);
      of[nf] = wmma_bf16(pf1, b1, of[nf]);
    }
    __syncthreads();
  }

  // finalize
#pragma unroll
  for (int nf = 0; nf < 4; ++nf)
#pragma unroll
    for (int i = 0; i < 8; ++i) {
      int lr = hh * 8 + i;
      int t = qb * 64 + w * 16 + lr;
      float val = of[nf][i] / lrow[w * 16 + lr];
      int hd = nf * 16 + rr;
      o_tok[((size_t)(b * NN + t)) * DD + h * HD + hd] = f2bf(val);
    }
}

// ---------------- host-side orchestration ----------------
extern "C" void kernel_launch(void* const* d_in, const int* in_sizes, int n_in,
                              void* d_out, int out_size, void* d_ws, size_t ws_size,
                              hipStream_t stream) {
  (void)in_sizes; (void)n_in; (void)out_size; (void)ws_size;
  const float* x   = (const float*)d_in[0];
  const float* g1  = (const float*)d_in[1];
  const float* b1  = (const float*)d_in[2];
  const float* g2  = (const float*)d_in[3];
  const float* b2  = (const float*)d_in[4];
  const float* gi  = (const float*)d_in[5];
  const float* bi  = (const float*)d_in[6];
  const float* Wg  = (const float*)d_in[7];
  const float* bg  = (const float*)d_in[8];
  const float* Wki = (const float*)d_in[9];
  const float* bki = (const float*)d_in[10];
  const float* Wvi = (const float*)d_in[11];
  const float* bvi = (const float*)d_in[12];
  const float* ema = (const float*)d_in[13];
  const float* Wq  = (const float*)d_in[14];
  const float* bq  = (const float*)d_in[15];
  const float* Wk  = (const float*)d_in[16];
  const float* bk  = (const float*)d_in[17];
  const float* Wv  = (const float*)d_in[18];
  const float* bv  = (const float*)d_in[19];
  const float* Wo  = (const float*)d_in[20];
  const float* bo  = (const float*)d_in[21];
  const float* W1  = (const float*)d_in[22];
  const float* b1f = (const float*)d_in[23];
  const float* W2  = (const float*)d_in[24];
  const float* b2f = (const float*)d_in[25];

  char* wsp = (char*)d_ws;
  auto alloc = [&](size_t bytes) -> void* {
    void* p = (void*)wsp;
    wsp += (bytes + 255) & ~(size_t)255;
    return p;
  };
  const size_t W_DD = (size_t)DD * DD;
  const size_t W_DF = (size_t)DD * FFN;
  const size_t ACT  = (size_t)MTOK * DD;
  const size_t ACTF = (size_t)MTOK * FFN;

  unsigned short* WgT  = (unsigned short*)alloc(W_DD * 2);
  unsigned short* WkiT = (unsigned short*)alloc(W_DD * 2);
  unsigned short* WviT = (unsigned short*)alloc(W_DD * 2);
  unsigned short* WqT  = (unsigned short*)alloc(W_DD * 2);
  unsigned short* WkT  = (unsigned short*)alloc(W_DD * 2);
  unsigned short* WvT  = (unsigned short*)alloc(W_DD * 2);
  unsigned short* WoT  = (unsigned short*)alloc(W_DD * 2);
  unsigned short* W1T  = (unsigned short*)alloc(W_DF * 2);
  unsigned short* W2T  = (unsigned short*)alloc(W_DF * 2);
  float*          xi_f = (float*)alloc(ACT * 4);
  float*          pool = (float*)alloc(ACT * 4);
  unsigned short* xi_b = (unsigned short*)alloc(ACT * 2);
  unsigned short* xn_b = (unsigned short*)alloc(ACT * 2);
  unsigned short* intb = (unsigned short*)alloc(ACT * 2);
  float*          kdel = (float*)alloc(ACT * 4);
  float*          vdel = (float*)alloc(ACT * 4);
  unsigned short* qh   = (unsigned short*)alloc(ACT * 2);
  unsigned short* kh   = (unsigned short*)alloc(ACT * 2);
  unsigned short* vhT  = (unsigned short*)alloc(ACT * 2);
  unsigned short* otok = (unsigned short*)alloc(ACT * 2);
  float*          x1   = (float*)alloc(ACT * 4);
  unsigned short* xn2b = (unsigned short*)alloc(ACT * 2);
  unsigned short* hbuf = (unsigned short*)alloc(ACTF * 2);

  // transposed bf16 weight conversions (WT[n][k])
  auto cvtT = [&](const float* s, unsigned short* dst, int K, int N) {
    int n = K * N;
    cvt_t_kernel<<<(n + 255) / 256, 256, 0, stream>>>(s, dst, K, N);
  };
  cvtT(Wg, WgT, DD, DD);  cvtT(Wki, WkiT, DD, DD); cvtT(Wvi, WviT, DD, DD);
  cvtT(Wq, WqT, DD, DD);  cvtT(Wk, WkT, DD, DD);   cvtT(Wv, WvT, DD, DD);
  cvtT(Wo, WoT, DD, DD);  cvtT(W1, W1T, DD, FFN);  cvtT(W2, W2T, FFN, DD);

  // LayerNorms on input
  ln_kernel<<<MTOK, 256, 0, stream>>>(x, gi, bi, xi_b, xi_f);
  ln_kernel<<<MTOK, 256, 0, stream>>>(x, g1, b1, xn_b, nullptr);

  // interference branch: EMA scan + AGC
  ema_kernel<<<(BB * DD) / 256, 256, 0, stream>>>(xi_f, ema, pool);
  agc_kernel<<<MTOK, 256, 0, stream>>>(pool);

  dim3 gD(DD / GBN, MTOK / GBM);
  dim3 gF1(FFN / GBN, MTOK / GBM);

  // inter = sigmoid(xi@Wg + bg) * pool
  gemm_bf16_kernel<1><<<gD, 256, 0, stream>>>(xi_b, WgT, bg, nullptr, pool, intb, nullptr, MTOK, DD, DD);
  // k_delta, v_delta (f32, token layout)
  gemm_bf16_kernel<2><<<gD, 256, 0, stream>>>(intb, WkiT, bki, nullptr, nullptr, nullptr, kdel, MTOK, DD, DD);
  gemm_bf16_kernel<2><<<gD, 256, 0, stream>>>(intb, WviT, bvi, nullptr, nullptr, nullptr, vdel, MTOK, DD, DD);
  // q (scaled), k (+k_delta) head layout; v (+v_delta) transposed head layout
  gemm_bf16_kernel<4><<<gD, 256, 0, stream>>>(xn_b, WqT, bq, nullptr, nullptr, qh, nullptr, MTOK, DD, DD);
  gemm_bf16_kernel<3><<<gD, 256, 0, stream>>>(xn_b, WkT, bk, kdel, nullptr, kh, nullptr, MTOK, DD, DD);
  gemm_bf16_kernel<7><<<gD, 256, 0, stream>>>(xn_b, WvT, bv, vdel, nullptr, vhT, nullptr, MTOK, DD, DD);

  // causal flash attention
  attn_kernel<<<dim3(NN / 64, HH, BB), 128, 0, stream>>>(qh, kh, vhT, otok);

  // x1 = x + o@Wo + bo
  gemm_bf16_kernel<5><<<gD, 256, 0, stream>>>(otok, WoT, bo, x, nullptr, nullptr, x1, MTOK, DD, DD);

  // FFN
  ln_kernel<<<MTOK, 256, 0, stream>>>(x1, g2, b2, xn2b, nullptr);
  gemm_bf16_kernel<6><<<gF1, 256, 0, stream>>>(xn2b, W1T, b1f, nullptr, nullptr, hbuf, nullptr, MTOK, DD, FFN);
  gemm_bf16_kernel<5><<<gD, 256, 0, stream>>>(hbuf, W2T, b2f, x1, nullptr, nullptr, (float*)d_out, MTOK, FFN, DD);
}
